// CGConv_45535243272312
// MI455X (gfx1250) — compile-verified, matrix operand order checked by reading, not run on hardware
//
#include <hip/hip_runtime.h>
#include <cstdint>
#include <cstddef>

// ---------------- problem constants ----------------
#define NATOMS 100000
#define MNBR   12
#define AF     64
#define BFEAT  41
#define KF     (2*AF + BFEAT)   // 169
#define KSTEPS 6                // K padded to 192 = 6 * 32
#define KPAD   (KSTEPS*32)      // 192
#define NCOLS  (2*AF)           // 128
#define NTILES (NCOLS/16)       // 8 col tiles -> 8 waves
#define G_ATOMS 4
#define ROWS   (G_ATOMS*MNBR)   // 48 rows per block
#define ROWT   (ROWS/16)        // 3 WMMA row tiles
#define NBLK   (NATOMS/G_ATOMS) // 25000
#define LDA    200              // LDS A row stride (bf16 elems) = 400B, 16B aligned, bank-friendly
#define LDR    132              // LDS result row stride (f32)
#define BN_EPS 1e-5f

typedef __attribute__((ext_vector_type(16))) __bf16 v16bf;
typedef __attribute__((ext_vector_type(8)))  float  v8f;

union BfPack { v16bf v; uint4 q[2]; };

__device__ __forceinline__ unsigned short f32_to_bf16(float f) {
    union { float f; uint32_t u; } x; x.f = f;
    uint32_t u = x.u;
    u += 0x7FFFu + ((u >> 16) & 1u);      // round-to-nearest-even
    return (unsigned short)(u >> 16);
}

__device__ __forceinline__ float softplus_f(float x) {
    return fmaxf(x, 0.f) + log1pf(__expf(-fabsf(x)));
}

// ---------------- kernel 0: pack W (169x128 f32) into per-lane WMMA-B bf16 layout ----------------
// Wpk index: ((nTile*KSTEPS + ks)*32 + lane)*16 + i ; lane<16 holds K {j,16+j}, lane>=16 holds {8+j,24+j}
__global__ void __launch_bounds__(256)
cg_prep_w(const float* __restrict__ W, unsigned short* __restrict__ Wpk) {
    int idx = blockIdx.x * 256 + threadIdx.x;
    if (idx >= NTILES * KSTEPS * 32 * 16) return;
    int i    = idx & 15;
    int lane = (idx >> 4) & 31;
    int ks   = (idx >> 9) % KSTEPS;
    int t    = idx / (16 * 32 * KSTEPS);
    int h = i >> 3, j = i & 7;
    int klocal = (lane & 16) ? (h ? 24 + j : 8 + j) : (h ? 16 + j : j);
    int k = ks * 32 + klocal;
    int n = t * 16 + (lane & 15);
    float val = (k < KF) ? W[(size_t)k * NCOLS + n] : 0.f;
    Wpk[idx] = f32_to_bf16(val);
}

// ---------------- shared A staging: 48 rows x 192 (bf16) ----------------
__device__ __forceinline__ void stage_A(unsigned short* lA, int* lNbr, int atom0,
                                        const int* __restrict__ nbr,
                                        const float* __restrict__ atomF,
                                        const float* __restrict__ bondF) {
    if (threadIdx.x < ROWS) {
        int r = threadIdx.x;
        lNbr[r] = nbr[(size_t)(atom0 + r / MNBR) * MNBR + (r % MNBR)];
    }
    __syncthreads();
    const int total = ROWS * KPAD;                 // 9216 elems
    for (int idx = threadIdx.x; idx < total; idx += 256) {
        int r = idx / KPAD;
        int k = idx - r * KPAD;
        int atom = atom0 + r / MNBR;
        int m = r % MNBR;
        float val;
        if (k < AF) {
            val = atomF[(size_t)atom * AF + k];
        } else if (k < 2 * AF) {
            val = atomF[(size_t)lNbr[r] * AF + (k - AF)];   // gather: L2-resident
        } else if (k < KF) {
            val = bondF[((size_t)atom * MNBR + m) * BFEAT + (k - 2 * AF)];
        } else {
            val = 0.f;
        }
        lA[r * LDA + k] = f32_to_bf16(val);
    }
}

// ---------------- WMMA core: 3 row-tiles x 1 col-tile per wave, K=192 ----------------
__device__ __forceinline__ void wmma_rows(const unsigned short* lA,
                                          const unsigned short* __restrict__ Wpk,
                                          int wv, int lane, v8f acc[ROWT]) {
    #pragma unroll
    for (int ks = 0; ks < KSTEPS; ++ks) {
        BfPack B;
        const uint4* bp = (const uint4*)(Wpk + ((size_t)(wv * KSTEPS + ks) * 32 + lane) * 16);
        B.q[0] = bp[0];                         // fully coalesced global_load_b128
        B.q[1] = bp[1];
        int koff0 = ks * 32 + ((lane & 16) ? 8 : 0);
        int koff1 = koff0 + 16;
        #pragma unroll
        for (int rt = 0; rt < ROWT; ++rt) {
            int row = rt * 16 + (lane & 15);
            BfPack A;
            A.q[0] = *(const uint4*)(lA + row * LDA + koff0);   // ds_load_b128
            A.q[1] = *(const uint4*)(lA + row * LDA + koff1);
            acc[rt] = __builtin_amdgcn_wmma_f32_16x16x32_bf16(
                false, A.v, false, B.v, (short)0, acc[rt], false, false);
        }
    }
}

// ---------------- pass 1: GEMM + per-block column sum / sumsq ----------------
__global__ void __launch_bounds__(256)
cg_pass1(const int* __restrict__ nbr, const float* __restrict__ atomF,
         const float* __restrict__ bondF, const unsigned short* __restrict__ Wpk,
         float* __restrict__ partial1) {
    __shared__ __align__(16) unsigned short lA[ROWS * LDA];
    __shared__ int lNbr[ROWS];
    int atom0 = blockIdx.x * G_ATOMS;
    stage_A(lA, lNbr, atom0, nbr, atomF, bondF);
    __syncthreads();

    int wv = threadIdx.x >> 5, lane = threadIdx.x & 31;
    v8f zero = {0.f,0.f,0.f,0.f,0.f,0.f,0.f,0.f};
    v8f acc[ROWT]; 
    #pragma unroll
    for (int rt = 0; rt < ROWT; ++rt) acc[rt] = zero;
    wmma_rows(lA, Wpk, wv, lane, acc);

    // stats: lane holds col n = wv*16 + lane%16; rows = rt*16 + v + 8*(lane>=16)
    float s = 0.f, q = 0.f;
    #pragma unroll
    for (int rt = 0; rt < ROWT; ++rt)
        #pragma unroll
        for (int v = 0; v < 8; ++v) { float x = acc[rt][v]; s += x; q += x * x; }
    s += __shfl_xor(s, 16, 32);
    q += __shfl_xor(q, 16, 32);
    if (lane < 16) {
        int c = wv * 16 + lane;
        partial1[(size_t)blockIdx.x * 256 + c]       = s;
        partial1[(size_t)blockIdx.x * 256 + 128 + c] = q;
    }
}

// ---------------- reduce 1: fold mean/var + bn1 params into per-col affine (bias b cancels in BN) ----
__global__ void __launch_bounds__(256)
cg_reduce1(const float* __restrict__ partial1, const float* __restrict__ bn1s,
           const float* __restrict__ bn1o, float* __restrict__ a1c1) {
    __shared__ float ls[256], lq[256];
    int c = blockIdx.x;                    // 0..127
    float s = 0.f, q = 0.f;
    for (int b = threadIdx.x; b < NBLK; b += 256) {
        s += partial1[(size_t)b * 256 + c];
        q += partial1[(size_t)b * 256 + 128 + c];
    }
    ls[threadIdx.x] = s; lq[threadIdx.x] = q;
    __syncthreads();
    for (int off = 128; off > 0; off >>= 1) {
        if ((int)threadIdx.x < off) { ls[threadIdx.x] += ls[threadIdx.x + off];
                                      lq[threadIdx.x] += lq[threadIdx.x + off]; }
        __syncthreads();
    }
    if (threadIdx.x == 0) {
        const float inv = 1.f / (float)(NATOMS * MNBR);
        float mean = ls[0] * inv;
        float var  = lq[0] * inv - mean * mean;
        float a = bn1s[c] * rsqrtf(var + BN_EPS);
        a1c1[c]       = a;
        a1c1[128 + c] = bn1o[c] - mean * a;
    }
}

// ---------------- pass 2: GEMM + BN1 affine + gate + neighbor sum + bn2 partials ----------------
__global__ void __launch_bounds__(256)
cg_pass2(const int* __restrict__ nbr, const float* __restrict__ atomF,
         const float* __restrict__ bondF, const unsigned short* __restrict__ Wpk,
         const float* __restrict__ a1c1, float* __restrict__ summed,
         float* __restrict__ partial2) {
    __shared__ __align__(16) unsigned short lA[ROWS * LDA];
    __shared__ int lNbr[ROWS];
    __shared__ float res[ROWS * LDR];
    int atom0 = blockIdx.x * G_ATOMS;
    stage_A(lA, lNbr, atom0, nbr, atomF, bondF);
    __syncthreads();

    int wv = threadIdx.x >> 5, lane = threadIdx.x & 31;
    v8f zero = {0.f,0.f,0.f,0.f,0.f,0.f,0.f,0.f};
    v8f acc[ROWT];
    #pragma unroll
    for (int rt = 0; rt < ROWT; ++rt) acc[rt] = zero;
    wmma_rows(lA, Wpk, wv, lane, acc);

    int n = wv * 16 + (lane & 15);
    float a1 = a1c1[n], c1 = a1c1[128 + n];
    int rofs = (lane & 16) ? 8 : 0;
    #pragma unroll
    for (int rt = 0; rt < ROWT; ++rt)
        #pragma unroll
        for (int v = 0; v < 8; ++v)
            res[(rt * 16 + v + rofs) * LDR + n] = acc[rt][v] * a1 + c1;
    __syncthreads();

    // per-atom gating + sum over 12 neighbors; thread = (atom g, col c)
    int g = threadIdx.x >> 6, c = threadIdx.x & 63;
    float sum = 0.f;
    #pragma unroll
    for (int m = 0; m < MNBR; ++m) {
        int row = g * MNBR + m;
        float f = res[row * LDR + c];
        float e = res[row * LDR + 64 + c];
        float sig = 1.f / (1.f + __expf(-f));
        sum += sig * softplus_f(e);
    }
    summed[((size_t)atom0 + g) * AF + c] = sum;

    __syncthreads();                      // done reading res; reuse as scratch
    res[threadIdx.x]       = sum;
    res[256 + threadIdx.x] = sum * sum;
    __syncthreads();
    if (threadIdx.x < 64) {
        int t = threadIdx.x;
        float S = res[t] + res[64 + t] + res[128 + t] + res[192 + t];
        float Q = res[256 + t] + res[320 + t] + res[384 + t] + res[448 + t];
        partial2[(size_t)blockIdx.x * 128 + t]      = S;
        partial2[(size_t)blockIdx.x * 128 + 64 + t] = Q;
    }
}

// ---------------- reduce 2 ----------------
__global__ void __launch_bounds__(256)
cg_reduce2(const float* __restrict__ partial2, const float* __restrict__ bn2s,
           const float* __restrict__ bn2o, float* __restrict__ a2c2) {
    __shared__ float ls[256], lq[256];
    int c = blockIdx.x;                    // 0..63
    float s = 0.f, q = 0.f;
    for (int b = threadIdx.x; b < NBLK; b += 256) {
        s += partial2[(size_t)b * 128 + c];
        q += partial2[(size_t)b * 128 + 64 + c];
    }
    ls[threadIdx.x] = s; lq[threadIdx.x] = q;
    __syncthreads();
    for (int off = 128; off > 0; off >>= 1) {
        if ((int)threadIdx.x < off) { ls[threadIdx.x] += ls[threadIdx.x + off];
                                      lq[threadIdx.x] += lq[threadIdx.x + off]; }
        __syncthreads();
    }
    if (threadIdx.x == 0) {
        const float inv = 1.f / (float)NATOMS;
        float mean = ls[0] * inv;
        float var  = lq[0] * inv - mean * mean;
        float a = bn2s[c] * rsqrtf(var + BN_EPS);
        a2c2[c]      = a;
        a2c2[64 + c] = bn2o[c] - mean * a;
    }
}

// ---------------- final elementwise ----------------
__global__ void __launch_bounds__(256)
cg_final(const float* __restrict__ atomF, const float* __restrict__ summed,
         const float* __restrict__ a2c2, float* __restrict__ out) {
    size_t idx = (size_t)blockIdx.x * 256 + threadIdx.x;
    if (idx >= (size_t)NATOMS * AF) return;
    int c = (int)(idx & 63);
    float y = summed[idx] * a2c2[c] + a2c2[64 + c];
    out[idx] = softplus_f(atomF[idx] + y);
}

// ---------------- launcher ----------------
extern "C" void kernel_launch(void* const* d_in, const int* in_sizes, int n_in,
                              void* d_out, int out_size, void* d_ws, size_t ws_size,
                              hipStream_t stream) {
    const int*   nbr   = (const int*)  d_in[0];
    const float* atomF = (const float*)d_in[1];
    const float* bondF = (const float*)d_in[2];
    const float* W     = (const float*)d_in[3];
    // d_in[4] = linear bias b : exactly cancelled by BN1 mean subtraction -> unused
    const float* bn1s  = (const float*)d_in[5];
    const float* bn1o  = (const float*)d_in[6];
    const float* bn2s  = (const float*)d_in[7];
    const float* bn2o  = (const float*)d_in[8];
    float* out = (float*)d_out;

    char* ws = (char*)d_ws;
    unsigned short* Wpk = (unsigned short*)ws;                               // 49152 B
    float* a1c1     = (float*)(ws + 49152);                                  // 1 KB
    float* a2c2     = (float*)(ws + 50176);                                  // 512 B
    float* partial1 = (float*)(ws + 51200);                                  // 25.6 MB
    float* partial2 = (float*)(ws + 51200 + (size_t)NBLK * 256 * 4);         // 12.8 MB
    float* summed   = (float*)(ws + 51200 + (size_t)NBLK * 256 * 4
                                         + (size_t)NBLK * 128 * 4);          // 25.6 MB

    cg_prep_w<<<(NTILES*KSTEPS*32*16 + 255) / 256, 256, 0, stream>>>(W, Wpk);
    cg_pass1 <<<NBLK, 256, 0, stream>>>(nbr, atomF, bondF, Wpk, partial1);
    cg_reduce1<<<NCOLS, 256, 0, stream>>>(partial1, bn1s, bn1o, a1c1);
    cg_pass2 <<<NBLK, 256, 0, stream>>>(nbr, atomF, bondF, Wpk, a1c1, summed, partial2);
    cg_reduce2<<<AF, 256, 0, stream>>>(partial2, bn2s, bn2o, a2c2);
    cg_final <<<(NATOMS*AF + 255) / 256, 256, 0, stream>>>(atomF, summed, a2c2, out);
}
